// MLP_NW_dist_2_att_32418413150560
// MI455X (gfx1250) — matrix operand order repaired and checked by reading, not compiled, not run on hardware
//
#include <hip/hip_runtime.h>

typedef __attribute__((ext_vector_type(16))) _Float16 v16h;
typedef __attribute__((ext_vector_type(8)))  float    v8f;

#define HD   64   // feature dim H
#define NT   32   // keys per tile (WMMA K=32 for f16)
#define WPB  4    // waves per block
#define MT   16   // queries per wave (WMMA M=16)
#define KPAD 68   // padded K/Q row stride (floats) -> conflict-free banks

// gfx1250 async copy: global -> LDS, 16 bytes per lane, tracked by ASYNCcnt.
__device__ __forceinline__ void async_b128(unsigned int lds_off, const void* g)
{
    asm volatile("global_load_async_to_lds_b128 %0, %1, off"
                 :: "v"(lds_off), "v"((unsigned long long)(uintptr_t)g)
                 : "memory");
}
__device__ __forceinline__ void wait_async0()
{
    asm volatile("s_wait_asynccnt 0x0" ::: "memory");
}

// ---------------------------------------------------------------------------
// Tiny MLP: Linear(din->64) -> ReLU -> Linear(64->64). One thread per row.
// Weights stored [in, out]. Output row-major [row][h]. Chunked accumulators
// keep VGPR usage low.
// ---------------------------------------------------------------------------
__global__ void __launch_bounds__(128)
mlp2_f32(const float* __restrict__ X, const float* __restrict__ W1,
         const float* __restrict__ B1, const float* __restrict__ W2,
         const float* __restrict__ B2, float* __restrict__ out,
         int rows, int din)
{
    __shared__ float sW2[HD * HD];
    __shared__ float sW1[3 * HD];
    __shared__ float sB1[HD], sB2[HD];
    for (int i = threadIdx.x; i < HD * HD; i += blockDim.x) sW2[i] = W2[i];
    for (int i = threadIdx.x; i < din * HD; i += blockDim.x) sW1[i] = W1[i];
    for (int i = threadIdx.x; i < HD; i += blockDim.x) { sB1[i] = B1[i]; sB2[i] = B2[i]; }
    __syncthreads();

    int row = blockIdx.x * blockDim.x + threadIdx.x;
    if (row >= rows) return;

    float x[3] = {0.f, 0.f, 0.f};
    for (int d = 0; d < din; ++d) x[d] = X[(size_t)row * din + d];

    float hid[HD];
    #pragma unroll
    for (int h = 0; h < HD; ++h) {
        float v = sB1[h];
        for (int d = 0; d < din; ++d) v += x[d] * sW1[d * HD + h];
        hid[h] = fmaxf(v, 0.f);
    }

    float* op = out + (size_t)row * HD;
    #pragma unroll 1
    for (int hc = 0; hc < 4; ++hc) {
        float acc[16];
        #pragma unroll
        for (int t = 0; t < 16; ++t) acc[t] = sB2[hc * 16 + t];
        #pragma unroll 4
        for (int j = 0; j < HD; ++j) {
            const float hj = hid[j];
            const float* wr = &sW2[j * HD + hc * 16];
            #pragma unroll
            for (int t = 0; t < 16; ++t) acc[t] += hj * wr[t];
        }
        #pragma unroll
        for (int t = 0; t < 16; ++t) op[hc * 16 + t] = acc[t];
    }
}

// ---------------------------------------------------------------------------
// V MLP (din=1), output f16 FEATURE-MAJOR [B][H][N] so attention can stage
// contiguous 32-key rows per feature for the WMMA B operand.
// ---------------------------------------------------------------------------
__global__ void __launch_bounds__(128)
mlpv_f16(const float* __restrict__ X, const float* __restrict__ W1,
         const float* __restrict__ B1, const float* __restrict__ W2,
         const float* __restrict__ B2, _Float16* __restrict__ outT,
         int rows, int Nn)
{
    __shared__ float sW2[HD * HD];
    __shared__ float sW1[HD];
    __shared__ float sB1[HD], sB2[HD];
    for (int i = threadIdx.x; i < HD * HD; i += blockDim.x) sW2[i] = W2[i];
    for (int i = threadIdx.x; i < HD; i += blockDim.x) { sW1[i] = W1[i]; sB1[i] = B1[i]; sB2[i] = B2[i]; }
    __syncthreads();

    int row = blockIdx.x * blockDim.x + threadIdx.x;
    if (row >= rows) return;
    int bb = row / Nn;
    int n  = row % Nn;

    float x0 = X[row];  // din == 1
    float hid[HD];
    #pragma unroll
    for (int h = 0; h < HD; ++h) hid[h] = fmaxf(sB1[h] + x0 * sW1[h], 0.f);

    #pragma unroll 1
    for (int hc = 0; hc < 4; ++hc) {
        float acc[16];
        #pragma unroll
        for (int t = 0; t < 16; ++t) acc[t] = sB2[hc * 16 + t];
        #pragma unroll 4
        for (int j = 0; j < HD; ++j) {
            const float hj = hid[j];
            const float* wr = &sW2[j * HD + hc * 16];
            #pragma unroll
            for (int t = 0; t < 16; ++t) acc[t] += hj * wr[t];
        }
        #pragma unroll
        for (int t = 0; t < 16; ++t)
            outT[((size_t)bb * HD + hc * 16 + t) * Nn + n] = (_Float16)acc[t];
    }
}

// ---------------------------------------------------------------------------
// Fused flash-style L1-distance attention with double-buffered async-to-LDS
// tile staging and WMMA f16 P*V accumulation.
//   score[n,m] = sum_h |K[n,h]-Q[m,h]|;  p = softmax_n(-score^2/2)
//   O[m,h]     = sum_n p[n,m] * V[n,h]   (v_wmma_f32_16x16x32_f16)
//   out[m]     = (sum_h O[m,h]*Wo[h]) / l[m] + bo
// ---------------------------------------------------------------------------
__global__ void __launch_bounds__(128)
attn_fused(const float* __restrict__ Kf, const float* __restrict__ Qf,
           const _Float16* __restrict__ Vh, const float* __restrict__ Wo,
           const float* __restrict__ bo, float* __restrict__ out,
           int Nn, int Mm)
{
    __shared__ __align__(16) float    sQ[WPB * MT][KPAD];   // 17 KB, padded rows
    __shared__ __align__(16) float    sK[2][NT][KPAD];      // 2 x 8.5 KB
    __shared__ __align__(32) _Float16 sVT[2][HD][NT];       // 2 x 4 KB, [h][n]

    const int tid  = threadIdx.x;
    const int bb   = blockIdx.y;
    const int lane = tid & 31;
    const int wave = tid >> 5;
    const int hi   = lane >> 4;        // wave half
    const int l16  = lane & 15;
    const int mbase = blockIdx.x * (WPB * MT) + wave * MT;
    const int qrow  = wave * MT + l16; // this lane's query row in sQ

    const float*    Kbase = Kf + (size_t)bb * Nn * HD;
    const _Float16* Vbase = Vh + (size_t)bb * HD * Nn;

    const unsigned int skOff = (unsigned int)(uintptr_t)&sK[0][0][0];
    const unsigned int svOff = (unsigned int)(uintptr_t)&sVT[0][0][0];
    const unsigned int skBuf = NT * KPAD * 4;    // bytes per K buffer
    const unsigned int svBuf = HD * NT * 2;      // bytes per V buffer

    // ---- Stage Q tile (64 rows) with regular loads; 2 threads per row. ----
    {
        const int row  = tid >> 1;               // 0..63
        const int half = tid & 1;                // 0..1
        const float* src = Qf + ((size_t)bb * Mm + mbase - wave * MT  // block base
                                 + row) * HD + half * 32;
        float* dst = &sQ[row][half * 32];
        #pragma unroll
        for (int c = 0; c < 8; ++c)
            *(float4*)(dst + c * 4) = *(const float4*)(src + c * 4);
    }

    // ---- Issue async loads for tile 0 into buffer 0. ----
    {
        #pragma unroll
        for (int it = 0; it < 4; ++it) {         // K: 32x64 f32 = 512 x 16B
            int i = tid + it * 128;
            async_b128(skOff + (unsigned)((i >> 4) * KPAD * 4 + (i & 15) * 16),
                       Kbase + (size_t)(i >> 4) * HD + (i & 15) * 4);
        }
        #pragma unroll
        for (int it = 0; it < 2; ++it) {         // V: 64x32 f16 = 256 x 16B
            int i = tid + it * 128;
            async_b128(svOff + (unsigned)(((i >> 2) * NT + (i & 3) * 8) * 2),
                       Vbase + (size_t)(i >> 2) * Nn + (i & 3) * 8);
        }
    }

    v8f Oacc[4] = {};
    float runm = -__builtin_huge_valf();
    float lsum = 0.f;
    const int koff = hi ? 8 : 0;

    for (int nb = 0; nb < Nn; nb += NT) {
        const int buf = (nb / NT) & 1;
        wait_async0();                 // current tile's async copies complete
        __syncthreads();               // all waves ready; prev compute done

        // ---- Prefetch next tile into the other buffer (overlaps compute). --
        if (nb + NT < Nn) {
            const int nbuf = buf ^ 1;
            const float*    ks = Kbase + (size_t)(nb + NT) * HD;
            const _Float16* vs = Vbase + (nb + NT);
            #pragma unroll
            for (int it = 0; it < 4; ++it) {
                int i = tid + it * 128;
                async_b128(skOff + nbuf * skBuf +
                           (unsigned)((i >> 4) * KPAD * 4 + (i & 15) * 16),
                           ks + (size_t)(i >> 4) * HD + (i & 15) * 4);
            }
            #pragma unroll
            for (int it = 0; it < 2; ++it) {
                int i = tid + it * 128;
                async_b128(svOff + nbuf * svBuf +
                           (unsigned)(((i >> 2) * NT + (i & 3) * 8) * 2),
                           vs + (size_t)(i >> 2) * Nn + (i & 3) * 8);
            }
        }

        // ---- L1 distances, h-outer to keep register pressure low. ----
        // Lane owns 16 keys in contiguous k order (same as B):
        // lane<16 -> K in {0..15}, lane>=16 -> {16..31}.
        float d[16];
        #pragma unroll
        for (int kk = 0; kk < 16; ++kk) d[kk] = 0.f;

        #pragma unroll 4
        for (int h4 = 0; h4 < HD / 4; ++h4) {
            const float4 qv = *(const float4*)(&sQ[qrow][h4 * 4]);
            #pragma unroll
            for (int kk = 0; kk < 16; ++kk) {
                const int ko = kk + 16 * hi;
                const float4 kv = *(const float4*)(&sK[buf][ko][h4 * 4]);
                d[kk] += fabsf(kv.x - qv.x) + fabsf(kv.y - qv.y) +
                         fabsf(kv.z - qv.z) + fabsf(kv.w - qv.w);
            }
        }

        float tmax = -__builtin_huge_valf();
        #pragma unroll
        for (int kk = 0; kk < 16; ++kk) {
            d[kk] = -0.5f * d[kk] * d[kk];       // score
            tmax  = fmaxf(tmax, d[kk]);
        }
        tmax = fmaxf(tmax, __shfl_xor(tmax, 16, 32));  // tile row max
        const float newm  = fmaxf(runm, tmax);
        const float alpha = __expf(runm - newm);
        runm = newm;

        float ps = 0.f;
        v16h a;
        #pragma unroll
        for (int kk = 0; kk < 16; ++kk) {
            const float p = __expf(d[kk] - newm);
            ps += p;
            a[kk] = (_Float16)p;                 // element kk == A-layout slot
        }
        lsum = lsum * alpha + ps;

        // Rescale accumulators: D-layout row of VGPR r is (r + 8*hi),
        // whose alpha lives in lane (r + 8*hi).
        #pragma unroll
        for (int r = 0; r < 8; ++r) {
            const float ar = __shfl(alpha, r + 8 * hi, 32);
            #pragma unroll
            for (int c = 0; c < 4; ++c) Oacc[c][r] *= ar;
        }

        // O += P(16x32) x V(32x16) per 16-feature chunk.
        #pragma unroll
        for (int c = 0; c < 4; ++c) {
            const v16h bmat = *(const v16h*)(&sVT[buf][c * 16 + l16][hi * 16]);
            Oacc[c] = __builtin_amdgcn_wmma_f32_16x16x32_f16(
                false, a, false, bmat, (short)0, Oacc[c], false, false);
        }
    }

    // ---- Epilogue: combine sums, project with Wo, write [B,M,1]. ----
    lsum += __shfl_xor(lsum, 16, 32);            // full l[m=l16] in every lane

    float wo[4];
    #pragma unroll
    for (int c = 0; c < 4; ++c) wo[c] = Wo[c * 16 + l16];
    const float bov = bo[0];

    float t[8];
    #pragma unroll
    for (int r = 0; r < 8; ++r) {
        float acc = 0.f;
        #pragma unroll
        for (int c = 0; c < 4; ++c) acc += Oacc[c][r] * wo[c];
        acc += __shfl_xor(acc, 1, 32);
        acc += __shfl_xor(acc, 2, 32);
        acc += __shfl_xor(acc, 4, 32);
        acc += __shfl_xor(acc, 8, 32);
        t[r] = acc;
    }
    float lr[8];
    #pragma unroll
    for (int r = 0; r < 8; ++r) lr[r] = __shfl(lsum, r + 8 * hi, 32);

    if (l16 == 0) {
        const int mb = mbase + 8 * hi;
        #pragma unroll
        for (int r = 0; r < 8; ++r)
            out[(size_t)bb * Mm + mb + r] = bov + t[r] / lr[r];
    }
}

// ---------------------------------------------------------------------------
extern "C" void kernel_launch(void* const* d_in, const int* in_sizes, int n_in,
                              void* d_out, int out_size, void* d_ws, size_t ws_size,
                              hipStream_t stream)
{
    (void)in_sizes; (void)n_in; (void)out_size; (void)ws_size;
    const int B = 4, N = 4096, M = 4096;

    const float* coords_f = (const float*)d_in[0];
    const float* values_f = (const float*)d_in[1];
    const float* coords_t = (const float*)d_in[2];
    const float* Wk1 = (const float*)d_in[3];
    const float* bk1 = (const float*)d_in[4];
    const float* Wk2 = (const float*)d_in[5];
    const float* bk2 = (const float*)d_in[6];
    const float* Wq1 = (const float*)d_in[7];
    const float* bq1 = (const float*)d_in[8];
    const float* Wq2 = (const float*)d_in[9];
    const float* bq2 = (const float*)d_in[10];
    const float* Wv1 = (const float*)d_in[11];
    const float* bv1 = (const float*)d_in[12];
    const float* Wv2 = (const float*)d_in[13];
    const float* bv2 = (const float*)d_in[14];
    const float* Wo  = (const float*)d_in[15];
    const float* bo  = (const float*)d_in[16];

    // Workspace: Kf (4 MB) | Qf (4 MB) | Vh (2 MB, feature-major f16)
    float*    Kf = (float*)d_ws;
    float*    Qf = Kf + (size_t)B * N * HD;
    _Float16* Vh = (_Float16*)(Qf + (size_t)B * M * HD);

    mlp2_f32<<<dim3((B * N + 127) / 128), 128, 0, stream>>>(
        coords_f, Wk1, bk1, Wk2, bk2, Kf, B * N, 3);
    mlp2_f32<<<dim3((B * M + 127) / 128), 128, 0, stream>>>(
        coords_t, Wq1, bq1, Wq2, bq2, Qf, B * M, 3);
    mlpv_f16<<<dim3((B * N + 127) / 128), 128, 0, stream>>>(
        values_f, Wv1, bv1, Wv2, bv2, Vh, B * N, N);

    attn_fused<<<dim3(M / (WPB * MT), B), 128, 0, stream>>>(
        Kf, Qf, Vh, Wo, bo, (float*)d_out, N, M);
}
